// MultiHeadCausalSelfAttention_13271448945022
// MI455X (gfx1250) — compile-verified, matrix operand order checked by reading, not run on hardware
//
#include <hip/hip_runtime.h>
#include <hip/hip_bf16.h>

#define T_SEQ   4096
#define D_MODEL 1024
#define NH      16
#define DH      64

typedef __attribute__((ext_vector_type(16))) __bf16 v16bf;
typedef __attribute__((ext_vector_type(8)))  float  v8f;

__device__ __forceinline__ unsigned short f2bf(float f) {
  unsigned int u = __float_as_uint(f);
  u += 0x7FFFu + ((u >> 16) & 1u);   // round-to-nearest-even
  return (unsigned short)(u >> 16);
}

__device__ __forceinline__ unsigned int pack2bf(float a, float b) {
  return (unsigned int)f2bf(a) | ((unsigned int)f2bf(b) << 16);
}

// CDNA5 16-bit A/B fragment: lane l holds row/col (l&15), with
// K = 8*(l>>4)+0..7 in elements 0..7 and K = 16+8*(l>>4)+0..7 in elements 8..15.
// p points at this lane's first 8 bf16; the second chunk is 16 elements later.
__device__ __forceinline__ v16bf load_frag(const unsigned short* p) {
  union { uint4 u[2]; v16bf v; } t;
  t.u[0] = *(const uint4*)(p);
  t.u[1] = *(const uint4*)(p + 16);
  return t.v;
}

__device__ __forceinline__ v8f wmma_bf16(v16bf a, v16bf b, v8f c) {
  return __builtin_amdgcn_wmma_f32_16x16x32_bf16(
      /*neg_a=*/false, a, /*neg_b=*/false, b,
      /*c_mod=*/(short)0, c, /*reuse_a=*/false, /*reuse_b=*/false);
}

__device__ __forceinline__ v8f zero8() {
  return (v8f){0.f,0.f,0.f,0.f,0.f,0.f,0.f,0.f};
}

// Vectorized fp32 -> bf16 conversion (n multiple of 4).
__global__ __launch_bounds__(256) void f32_to_bf16_kernel(
    const float* __restrict__ src, unsigned short* __restrict__ dst, int n) {
  int i = blockIdx.x * blockDim.x + threadIdx.x;
  int stride = gridDim.x * blockDim.x;
  const int n4 = n >> 2;
  for (; i < n4; i += stride) {
    float4 f = ((const float4*)src)[i];
    uint2 o;
    o.x = pack2bf(f.x, f.y);
    o.y = pack2bf(f.z, f.w);
    ((uint2*)dst)[i] = o;
  }
}

// C[M,N] = A[M,Kd] * W[N,Kd]^T   (i.e. torch Linear)
// mode 0: store bf16 row-major [M,N]     (Q, K)
// mode 1: store bf16 transposed [N,M]    (V -> Vt[feature][t], packed b128)
// mode 2: store fp32 [M,N] + bias        (output projection)
__global__ __launch_bounds__(256) void gemm_wmma_kernel(
    const unsigned short* __restrict__ A,
    const unsigned short* __restrict__ W,
    unsigned short* __restrict__ outB,
    float* __restrict__ outF,
    const float* __restrict__ bias,
    int M, int N, int Kd, int mode) {
  const int lane = threadIdx.x & 31;
  const int w    = threadIdx.x >> 5;
  const int m0   = blockIdx.y * 32  + (w & 1) * 16;
  const int n0   = blockIdx.x * 256 + (w >> 1) * 64;
  const int l16  = lane & 15;
  const int lhi  = (lane >> 4) << 3;   // 0 or 8

  v8f acc[4];
#pragma unroll
  for (int i = 0; i < 4; ++i) acc[i] = zero8();

  const unsigned short* ap = A + (size_t)(m0 + l16) * Kd + lhi;
  const unsigned short* wp = W + (size_t)(n0 + l16) * Kd + lhi;

  // Software pipeline: fragments for step k live while step k+1's loads are
  // in flight, so WMMAs only need partial s_wait_loadcnt.
  v16bf af = load_frag(ap);
  v16bf bf0 = load_frag(wp);
  v16bf bf1 = load_frag(wp + (size_t)16 * Kd);
  v16bf bf2 = load_frag(wp + (size_t)32 * Kd);
  v16bf bf3 = load_frag(wp + (size_t)48 * Kd);

  for (int k0 = 0; k0 < Kd - 32; k0 += 32) {
    const int kn = k0 + 32;
    // prefetch two steps ahead into cache (global_prefetch_b8, speculative)
    __builtin_prefetch(ap + kn + 32, 0, 3);
    __builtin_prefetch(wp + kn + 32, 0, 3);
    v16bf af_n  = load_frag(ap + kn);
    v16bf bf0_n = load_frag(wp + kn);
    v16bf bf1_n = load_frag(wp + (size_t)16 * Kd + kn);
    v16bf bf2_n = load_frag(wp + (size_t)32 * Kd + kn);
    v16bf bf3_n = load_frag(wp + (size_t)48 * Kd + kn);

    acc[0] = wmma_bf16(af, bf0, acc[0]);
    acc[1] = wmma_bf16(af, bf1, acc[1]);
    acc[2] = wmma_bf16(af, bf2, acc[2]);
    acc[3] = wmma_bf16(af, bf3, acc[3]);

    af = af_n; bf0 = bf0_n; bf1 = bf1_n; bf2 = bf2_n; bf3 = bf3_n;
  }
  acc[0] = wmma_bf16(af, bf0, acc[0]);
  acc[1] = wmma_bf16(af, bf1, acc[1]);
  acc[2] = wmma_bf16(af, bf2, acc[2]);
  acc[3] = wmma_bf16(af, bf3, acc[3]);

  if (mode == 1) {
    // Vt[feature][t]: the 8 accumulator rows are contiguous along t -> b128.
#pragma unroll
    for (int nt = 0; nt < 4; ++nt) {
      const int nc = n0 + nt * 16 + l16;
      uint4 st;
      st.x = pack2bf(acc[nt][0], acc[nt][1]);
      st.y = pack2bf(acc[nt][2], acc[nt][3]);
      st.z = pack2bf(acc[nt][4], acc[nt][5]);
      st.w = pack2bf(acc[nt][6], acc[nt][7]);
      *(uint4*)&outB[(size_t)nc * M + m0 + lhi] = st;
    }
  } else {
#pragma unroll
    for (int nt = 0; nt < 4; ++nt) {
      const int nc = n0 + nt * 16 + l16;
#pragma unroll
      for (int r = 0; r < 8; ++r) {
        const int mr = m0 + r + lhi;           // C/D layout: row = r + 8*(lane>>4)
        const float v = acc[nt][r];
        if (mode == 0) outB[(size_t)mr * N + nc] = f2bf(v);
        else           outF[(size_t)mr * N + nc] = v + bias[nc];
      }
    }
  }
}

// One wave handles one (head, 16-row q-tile). 8 waves per block.
__global__ __launch_bounds__(256) void flash_attn_kernel(
    const unsigned short* __restrict__ Qb,  // [T][1024] bf16
    const unsigned short* __restrict__ Kb,  // [T][1024] bf16
    const unsigned short* __restrict__ Vt,  // [1024][T] bf16 (transposed V)
    unsigned short* __restrict__ Zb) {      // [T][1024] bf16
  __shared__ __align__(16) unsigned short plds[8][16][32];  // per-wave P tile

  const int lane = threadIdx.x & 31;
  const int w    = threadIdx.x >> 5;
  const int wid  = blockIdx.x * 8 + w;
  const int head = wid >> 8;               // T/16 = 256 q-tiles per head
  const int q0   = (wid & 255) * 16;
  const int l16  = lane & 15;
  const int lhi  = (lane >> 4) << 3;

  // Q fragments: rows q0..q0+15, K = head's 64 features (two 32-chunks)
  const unsigned short* qp = Qb + (size_t)(q0 + l16) * D_MODEL + head * DH + lhi;
  const v16bf qf0 = load_frag(qp);
  const v16bf qf1 = load_frag(qp + 32);

  float mrow[8], lrow[8];
  v8f o[4];
#pragma unroll
  for (int r = 0; r < 8; ++r) { mrow[r] = -1e30f; lrow[r] = 0.f; }
#pragma unroll
  for (int i = 0; i < 4; ++i) o[i] = zero8();

  const float sc = 0.125f;  // 1/sqrt(64)

  for (int j0 = 0; j0 <= q0 + 15; j0 += 32) {
    // ---- batch all operand loads for this chunk up front ----
    const unsigned short* kp0 = Kb + (size_t)(j0 + l16) * D_MODEL + head * DH + lhi;
    const unsigned short* kp1 = kp0 + (size_t)16 * D_MODEL;
    __builtin_prefetch(kp0 + (size_t)32 * D_MODEL, 0, 3);  // next chunk's K rows
    v16bf kf00 = load_frag(kp0);
    v16bf kf01 = load_frag(kp0 + 32);
    v16bf kf10 = load_frag(kp1);
    v16bf kf11 = load_frag(kp1 + 32);
    v16bf vf[4];
#pragma unroll
    for (int nt = 0; nt < 4; ++nt)
      vf[nt] = load_frag(Vt + (size_t)(head * DH + nt * 16 + l16) * T_SEQ + j0 + lhi);

    // ---- S = Q * K^T for 32 kv columns (two 16x16 tiles, K=64 chained) ----
    v8f s0 = zero8(), s1 = zero8();
    s0 = wmma_bf16(qf0, kf00, s0);
    s0 = wmma_bf16(qf1, kf01, s0);
    s1 = wmma_bf16(qf0, kf10, s1);
    s1 = wmma_bf16(qf1, kf11, s1);

    // ---- scale + causal mask + online softmax (fp32, in C layout) ----
    // V-fragment loads above complete under this VALU work.
    const int c0 = j0 + l16;
    const int c1 = c0 + 16;
#pragma unroll
    for (int r = 0; r < 8; ++r) {
      const int row = q0 + r + lhi;
      float a = s0[r] * sc; if (c0 > row) a = -1e30f;
      float b = s1[r] * sc; if (c1 > row) b = -1e30f;
      // row max across the 16 lanes of this half-wave
      float v = fmaxf(a, b);
      v = fmaxf(v, __shfl_xor(v, 1, 32));
      v = fmaxf(v, __shfl_xor(v, 2, 32));
      v = fmaxf(v, __shfl_xor(v, 4, 32));
      v = fmaxf(v, __shfl_xor(v, 8, 32));
      const float mnew = fmaxf(mrow[r], v);
      const float corr = __expf(mrow[r] - mnew);
      mrow[r] = mnew;
      const float p0 = __expf(a - mnew);
      const float p1 = __expf(b - mnew);
      float ps = p0 + p1;
      ps += __shfl_xor(ps, 1, 32);
      ps += __shfl_xor(ps, 2, 32);
      ps += __shfl_xor(ps, 4, 32);
      ps += __shfl_xor(ps, 8, 32);
      lrow[r] = lrow[r] * corr + ps;
      o[0][r] *= corr; o[1][r] *= corr; o[2][r] *= corr; o[3][r] *= corr;
      // stage P (bf16) into this wave's LDS tile for re-layout into an A-fragment
      plds[w][r + lhi][l16]      = f2bf(p0);
      plds[w][r + lhi][l16 + 16] = f2bf(p1);
    }

    // ---- O += P * V (P: 16x32 A-fragment from LDS; V pre-loaded) ----
    v16bf pf = load_frag(&plds[w][l16][lhi]);
    o[0] = wmma_bf16(pf, vf[0], o[0]);
    o[1] = wmma_bf16(pf, vf[1], o[1]);
    o[2] = wmma_bf16(pf, vf[2], o[2]);
    o[3] = wmma_bf16(pf, vf[3], o[3]);
  }

  // ---- normalize and store Z (bf16) ----
#pragma unroll
  for (int r = 0; r < 8; ++r) {
    const float inv = 1.0f / lrow[r];
    const size_t rowoff = (size_t)(q0 + r + lhi) * D_MODEL + head * DH;
#pragma unroll
    for (int nt = 0; nt < 4; ++nt)
      Zb[rowoff + nt * 16 + l16] = f2bf(o[nt][r] * inv);
  }
}

extern "C" void kernel_launch(void* const* d_in, const int* in_sizes, int n_in,
                              void* d_out, int out_size, void* d_ws, size_t ws_size,
                              hipStream_t stream) {
  const float* x  = (const float*)d_in[0];
  const float* Wq = (const float*)d_in[1];
  const float* Wk = (const float*)d_in[2];
  const float* Wv = (const float*)d_in[3];
  const float* Wo = (const float*)d_in[4];
  const float* bo = (const float*)d_in[5];

  unsigned char* ws = (unsigned char*)d_ws;
  const size_t MB = 1u << 20;
  unsigned short* xb  = (unsigned short*)(ws + 0 * MB);   // 8 MB
  unsigned short* wqb = (unsigned short*)(ws + 8 * MB);   // 2 MB
  unsigned short* wkb = (unsigned short*)(ws + 10 * MB);
  unsigned short* wvb = (unsigned short*)(ws + 12 * MB);
  unsigned short* wob = (unsigned short*)(ws + 14 * MB);
  unsigned short* Qb  = (unsigned short*)(ws + 16 * MB);  // 8 MB
  unsigned short* Kb  = (unsigned short*)(ws + 24 * MB);  // 8 MB
  unsigned short* Vt  = (unsigned short*)(ws + 32 * MB);  // 8 MB, [1024][4096]
  unsigned short* Zb  = (unsigned short*)(ws + 40 * MB);  // 8 MB

  const int nX = T_SEQ * D_MODEL;
  const int nW = D_MODEL * D_MODEL;
  f32_to_bf16_kernel<<<1024, 256, 0, stream>>>(x,  xb,  nX);
  f32_to_bf16_kernel<<<512,  256, 0, stream>>>(Wq, wqb, nW);
  f32_to_bf16_kernel<<<512,  256, 0, stream>>>(Wk, wkb, nW);
  f32_to_bf16_kernel<<<512,  256, 0, stream>>>(Wv, wvb, nW);
  f32_to_bf16_kernel<<<512,  256, 0, stream>>>(Wo, wob, nW);

  dim3 gg(D_MODEL / 256, T_SEQ / 32);  // (4, 128)
  gemm_wmma_kernel<<<gg, 256, 0, stream>>>(xb, wqb, Qb, nullptr, nullptr,
                                           T_SEQ, D_MODEL, D_MODEL, 0);
  gemm_wmma_kernel<<<gg, 256, 0, stream>>>(xb, wkb, Kb, nullptr, nullptr,
                                           T_SEQ, D_MODEL, D_MODEL, 0);
  gemm_wmma_kernel<<<gg, 256, 0, stream>>>(xb, wvb, Vt, nullptr, nullptr,
                                           T_SEQ, D_MODEL, D_MODEL, 1);

  flash_attn_kernel<<<(NH * (T_SEQ / 16)) / 8, 256, 0, stream>>>(Qb, Kb, Vt, Zb);

  gemm_wmma_kernel<<<gg, 256, 0, stream>>>(Zb, wob, nullptr, (float*)d_out, bo,
                                           T_SEQ, D_MODEL, D_MODEL, 2);
}
